// MultiHeadAttention_14723147890835
// MI455X (gfx1250) — matrix operand order repaired and checked by reading, not compiled, not run on hardware
//
#include <hip/hip_runtime.h>

// ---------------------------------------------------------------------------
// Types
// ---------------------------------------------------------------------------
typedef __bf16 bf16;
typedef bf16  v4bf  __attribute__((ext_vector_type(4)));
typedef bf16  v8bf  __attribute__((ext_vector_type(8)));
typedef bf16  v16bf __attribute__((ext_vector_type(16)));
typedef float v8f   __attribute__((ext_vector_type(8)));
typedef unsigned int u32x4 __attribute__((ext_vector_type(4)));
typedef int          i32x4 __attribute__((ext_vector_type(4)));
typedef int          i32x8 __attribute__((ext_vector_type(8)));

#define Bq   4
#define Sq   2048
#define Dq   1024
#define Hq   16
#define HDq  64

#if defined(__gfx1250__) && __has_builtin(__builtin_amdgcn_tensor_load_to_lds)
#define HAVE_TDM 1
#else
#define HAVE_TDM 0
#endif

__device__ inline v8f zero8() {
  v8f z;
#pragma unroll
  for (int i = 0; i < 8; ++i) z[i] = 0.0f;
  return z;
}

// D = A(16x32 bf16) * B(32x16 bf16) + C (16x16 f32)
__device__ inline v8f wmma_bf16(v16bf a, v16bf b, v8f c) {
  return __builtin_amdgcn_wmma_f32_16x16x32_bf16(
      false, a, false, b, (short)0, c, false, false);
}

// A-fragment (16x32, MxK) from LDS row-major [16][stride].
// lane L<16: M=L, VGPR0-3=K0..7, VGPR4-7=K16..23; lane L+16: K8..15 / K24..31.
__device__ inline v16bf frag_a(const bf16* p, int stride, int lane, int k0) {
  const bf16* q = p + (lane & 15) * stride + k0 + (lane >> 4) * 8;
  v16bf f;
  *(v8bf*)&f       = *(const v8bf*)(q);
  *((v8bf*)&f + 1) = *(const v8bf*)(q + 16);
  return f;
}

// B-fragment (32x16, KxN) from LDS stored [N][K].
// lane L<16: N=L, K0..7 / K16..23; lane L+16: N=L, K8..15 / K24..31.
__device__ inline v16bf frag_b(const bf16* p, int stride, int lane, int k0) {
  const bf16* q = p + (lane & 15) * stride + k0 + (lane >> 4) * 8;
  v16bf f;
  *(v8bf*)&f       = *(const v8bf*)(q);
  *((v8bf*)&f + 1) = *(const v8bf*)(q + 16);
  return f;
}

__device__ inline float rmax16(float v) {
#pragma unroll
  for (int m = 1; m < 16; m <<= 1) v = fmaxf(v, __shfl_xor(v, m, 32));
  return v;
}
__device__ inline float rsum16(float v) {
#pragma unroll
  for (int m = 1; m < 16; m <<= 1) v += __shfl_xor(v, m, 32);
  return v;
}

// ---------------------------------------------------------------------------
// TDM: DMA a 2D bf16 tile (tile_w x tile_h elems, row stride in elems) from
// global memory into LDS (rows stored contiguously). ISA ch.8 D# layout:
//  g0: [1:0]count=1 | [63:32]lds_addr | [120:64]global_addr | [127:126]type=2
//  g1: [17:16]data_size=1(2B) | dims/strides packed as below.
// Issued by one wave (EXEC ignored by TDM); tracked by TENSORcnt.
// ---------------------------------------------------------------------------
#if HAVE_TDM
__device__ inline void tdm_load_2d_bf16(unsigned lds_off, const bf16* gaddr,
                                        unsigned tile_w, unsigned tile_h,
                                        unsigned stride_elems) {
  unsigned long long ga = (unsigned long long)(uintptr_t)gaddr;
  u32x4 g0;
  g0[0] = 1u;                                         // count=1, user mode
  g0[1] = lds_off;                                    // lds_addr (bytes)
  g0[2] = (unsigned)(ga & 0xffffffffu);               // global_addr[31:0]
  g0[3] = (unsigned)((ga >> 32) & 0x01ffffffu) | (2u << 30);  // [56:32]|type=2
  i32x8 g1;
  g1[0] = (int)(1u << 16);                            // data_size=1 (2 bytes)
  g1[1] = (int)((tile_w & 0xffffu) << 16);            // tensor_dim0[15:0]
  g1[2] = (int)(((tile_w >> 16) & 0xffffu) |          // tensor_dim0[31:16]
                ((tile_h & 0xffffu) << 16));          // tensor_dim1[15:0]
  g1[3] = (int)(((tile_h >> 16) & 0xffffu) |          // tensor_dim1[31:16]
                ((tile_w & 0xffffu) << 16));          // tile_dim0
  g1[4] = (int)(tile_h & 0xffffu);                    // tile_dim1 (tile_dim2=0)
  g1[5] = (int)stride_elems;                          // tensor_dim0_stride lo
  g1[6] = 0;                                          // stride hi, dim1_stride
  g1[7] = 0;
  i32x4 z4 = {0, 0, 0, 0};                            // groups 2/3 unused (2D)
  i32x8 z8 = {0, 0, 0, 0, 0, 0, 0, 0};
  __builtin_amdgcn_tensor_load_to_lds(g0, g1, z4, z4, z8, 0);
}
__device__ inline unsigned lds_off_of(const void* p) {
  return (unsigned)(uintptr_t)p;   // generic LDS addr: low 32 bits = offset
}
#endif

// Retire TDM ops down to PENDING (constant immediate required by SOPP),
// then make the tile visible to all waves.
template <int PENDING>
__device__ inline void tdm_commit(int t) {
#if HAVE_TDM
  if (t < 32) __builtin_amdgcn_s_wait_tensorcnt(PENDING);
#endif
  __syncthreads();
}

// ---------------------------------------------------------------------------
// One-shot fp32 -> bf16 conversion kernels (feed the TDM raw-bf16 path).
// ---------------------------------------------------------------------------
__global__ __launch_bounds__(256) void cvt_f32_bf16(
    const float* __restrict__ in, bf16* __restrict__ out, int n) {
  int i = (blockIdx.x * 256 + threadIdx.x) * 4;
  if (i + 3 < n) {
    float4 v = *(const float4*)(in + i);
    v4bf o;
    o[0] = (bf16)v.x; o[1] = (bf16)v.y; o[2] = (bf16)v.z; o[3] = (bf16)v.w;
    *(v4bf*)(out + i) = o;
  }
}

// out[N,K] = (bf16) in[K,N]^T  (LDS-tiled, coalesced both ways)
__global__ __launch_bounds__(256) void cvt_transpose_bf16(
    const float* __restrict__ in, bf16* __restrict__ out, int K, int N) {
  __shared__ float tile[32][33];
  const int kt = blockIdx.y * 32, nt = blockIdx.x * 32;
  const int tx = threadIdx.x & 31, ty = threadIdx.x >> 5;  // 32 x 8
#pragma unroll
  for (int r = ty; r < 32; r += 8)
    tile[r][tx] = in[(size_t)(kt + r) * N + nt + tx];
  __syncthreads();
#pragma unroll
  for (int r = ty; r < 32; r += 8)
    out[(size_t)(nt + r) * K + kt + tx] = (bf16)tile[tx][r];
}

// ---------------------------------------------------------------------------
// GEMM: C[M,N] = A[M,K] * Wt[N,K]^T + bias[N]   (A, Wt bf16; fp32 accum)
// Block tile 128x128, K staged 32/step, double-buffered TDM. 8 waves; each
// wave computes 32x64 (2 A-frags x 4 B-frags = 8 WMMA per stage).
// ---------------------------------------------------------------------------
template <typename TO>
__global__ __launch_bounds__(256) void gemm_bias_kernel(
    const bf16* __restrict__ A, const bf16* __restrict__ Wt,
    const float* __restrict__ bias, TO* __restrict__ C,
    int M, int N, int K) {
  __shared__ bf16 As[2][128 * 32];   // [row][k]
  __shared__ bf16 Bs[2][128 * 32];   // [n][k]

  const int t    = threadIdx.x;
  const int lane = t & 31;
  const int wid  = t >> 5;
  const int wm   = wid >> 1;         // 0..3 -> rows wm*32
  const int wn   = wid & 1;          // 0..1 -> cols wn*64
  const int m0   = blockIdx.y * 128;
  const int n0   = blockIdx.x * 128;
  const int nst  = K / 32;

  auto stage = [&](int buf, int k0) {
#if HAVE_TDM
    if (t < 32) {
      tdm_load_2d_bf16(lds_off_of(&As[buf][0]), A + (size_t)m0 * K + k0,
                       32, 128, (unsigned)K);
      tdm_load_2d_bf16(lds_off_of(&Bs[buf][0]), Wt + (size_t)n0 * K + k0,
                       32, 128, (unsigned)K);
    }
#else
#pragma unroll
    for (int i = t; i < 128 * 32; i += 256) {
      int r = i >> 5, c = i & 31;
      As[buf][i] = A[(size_t)(m0 + r) * K + k0 + c];
      Bs[buf][i] = Wt[(size_t)(n0 + r) * K + k0 + c];
    }
#endif
  };

  v8f acc[2][4];
#pragma unroll
  for (int i = 0; i < 2; ++i)
#pragma unroll
    for (int j = 0; j < 4; ++j) acc[i][j] = zero8();

  stage(0, 0);
  for (int s = 0; s < nst; ++s) {
    const int cur = s & 1;
    if (s + 1 < nst) {
      stage(cur ^ 1, (s + 1) * 32);  // prefetch next stage (2 TDM ops)
      tdm_commit<2>(t);              // retire this stage's 2 TDM ops
    } else {
      tdm_commit<0>(t);              // last stage: drain
    }

    v16bf a0 = frag_a(&As[cur][(wm * 32 + 0) * 32], 32, lane, 0);
    v16bf a1 = frag_a(&As[cur][(wm * 32 + 16) * 32], 32, lane, 0);
#pragma unroll
    for (int nt = 0; nt < 4; ++nt) {
      v16bf bfr = frag_b(&Bs[cur][(wn * 64 + nt * 16) * 32], 32, lane, 0);
      acc[0][nt] = wmma_bf16(a0, bfr, acc[0][nt]);
      acc[1][nt] = wmma_bf16(a1, bfr, acc[1][nt]);
    }
    __syncthreads();                 // done reading buf before re-fill
  }

  const int colb = lane & 15;
  const int hf   = lane >> 4;
#pragma unroll
  for (int mi = 0; mi < 2; ++mi)
#pragma unroll
    for (int nt = 0; nt < 4; ++nt) {
      int col = n0 + wn * 64 + nt * 16 + colb;
      float bv = bias[col];
#pragma unroll
      for (int r = 0; r < 8; ++r) {
        int row = m0 + wm * 32 + mi * 16 + hf * 8 + r;
        C[(size_t)row * N + col] = (TO)(acc[mi][nt][r] + bv);
      }
    }
}

// ---------------------------------------------------------------------------
// Flash-style attention over bf16 qkv ([B,S,3D] packed q|k|v).
// 128 threads (4 waves) own (b, h, 64 query rows); wave owns 16 rows.
// Q and K tiles arrive via TDM; V is transpose-staged (vectorized).
// 1/sqrt(HD) applied to fp32 scores post-WMMA.
// ---------------------------------------------------------------------------
__global__ __launch_bounds__(128) void attn_kernel(
    const bf16* __restrict__ qkv, bf16* __restrict__ o) {
  __shared__ bf16 sQ[64 * 64];    // [qrow][dim]
  __shared__ bf16 sK[64 * 64];    // [key][dim] == [N][K] for QK^T
  __shared__ bf16 sVt[64 * 64];   // [dim][key] == [N][K] for PV
  __shared__ bf16 sP[64 * 64];    // [qrow][key]

  const int t    = threadIdx.x;
  const int lane = t & 31;
  const int w    = t >> 5;
  const int q0   = blockIdx.x * 64;
  const int b    = blockIdx.y >> 4;
  const int h    = blockIdx.y & 15;

  const size_t rstride = 3 * Dq;
  const bf16* base = qkv + (size_t)(b * Sq) * rstride + h * HDq;

  // ---- stage Q (raw bf16) ----
#if HAVE_TDM
  if (t < 32)
    tdm_load_2d_bf16(lds_off_of(&sQ[0]), base + (size_t)q0 * rstride,
                     64, 64, (unsigned)rstride);
#else
#pragma unroll
  for (int c = t; c < 512; c += 128) {
    int r = c >> 3, d8 = (c & 7) * 8;
    *(v8bf*)&sQ[r * 64 + d8] =
        *(const v8bf*)(base + (size_t)(q0 + r) * rstride + d8);
  }
#endif
  tdm_commit<0>(t);

  v16bf aq0 = frag_a(sQ + w * 16 * 64, 64, lane, 0);
  v16bf aq1 = frag_a(sQ + w * 16 * 64, 64, lane, 32);

  v8f oacc[4];
  float mrun[8], lrun[8];
#pragma unroll
  for (int i = 0; i < 4; ++i) oacc[i] = zero8();
#pragma unroll
  for (int r = 0; r < 8; ++r) { mrun[r] = -3.0e38f; lrun[r] = 0.0f; }

  for (int j = 0; j < Sq / 64; ++j) {
    const int k0s = j * 64;
    // K tile via TDM (raw copy), V tile transposed by hand (vectorized).
#if HAVE_TDM
    if (t < 32)
      tdm_load_2d_bf16(lds_off_of(&sK[0]),
                       base + (size_t)k0s * rstride + Dq,
                       64, 64, (unsigned)rstride);
#else
#pragma unroll
    for (int c = t; c < 512; c += 128) {
      int r = c >> 3, d8 = (c & 7) * 8;
      *(v8bf*)&sK[r * 64 + d8] =
          *(const v8bf*)(base + (size_t)(k0s + r) * rstride + Dq + d8);
    }
#endif
#pragma unroll
    for (int c = t; c < 512; c += 128) {
      int r = c >> 3, d8 = (c & 7) * 8;
      v8bf vv = *(const v8bf*)(base + (size_t)(k0s + r) * rstride + 2 * Dq + d8);
#pragma unroll
      for (int e = 0; e < 8; ++e) sVt[(d8 + e) * 64 + r] = vv[e];
    }
    tdm_commit<0>(t);

    // S = Q K^T (fp32 accum), then scale by 1/sqrt(HD)
    v8f sacc[4];
#pragma unroll
    for (int kt = 0; kt < 4; ++kt) {
      v16bf bk0 = frag_b(sK + kt * 16 * 64, 64, lane, 0);
      v16bf bk1 = frag_b(sK + kt * 16 * 64, 64, lane, 32);
      sacc[kt] = wmma_bf16(aq0, bk0, zero8());
      sacc[kt] = wmma_bf16(aq1, bk1, sacc[kt]);
    }
#pragma unroll
    for (int kt = 0; kt < 4; ++kt)
#pragma unroll
      for (int r = 0; r < 8; ++r) sacc[kt][r] *= 0.125f;

    // Online softmax (per-wave rows; 16-lane xor reductions per half)
#pragma unroll
    for (int r = 0; r < 8; ++r) {
      float mx = fmaxf(fmaxf(sacc[0][r], sacc[1][r]),
                       fmaxf(sacc[2][r], sacc[3][r]));
      mx = rmax16(mx);
      float mn   = fmaxf(mrun[r], mx);
      float corr = __expf(mrun[r] - mn);
      mrun[r] = mn;
      float rs = 0.0f;
#pragma unroll
      for (int kt = 0; kt < 4; ++kt) {
        float p = __expf(sacc[kt][r] - mn);
        sacc[kt][r] = p;
        rs += p;
      }
      rs = rsum16(rs);
      lrun[r] = lrun[r] * corr + rs;
#pragma unroll
      for (int ot = 0; ot < 4; ++ot) oacc[ot][r] *= corr;
    }

    // Re-layout P through LDS into A-fragment form
#pragma unroll
    for (int kt = 0; kt < 4; ++kt)
#pragma unroll
      for (int r = 0; r < 8; ++r)
        sP[(w * 16 + (lane >> 4) * 8 + r) * 64 + kt * 16 + (lane & 15)] =
            (bf16)sacc[kt][r];
    __syncthreads();

    // O += P V
    v16bf ap0 = frag_a(sP + w * 16 * 64, 64, lane, 0);
    v16bf ap1 = frag_a(sP + w * 16 * 64, 64, lane, 32);
#pragma unroll
    for (int ot = 0; ot < 4; ++ot) {
      v16bf bv0 = frag_b(sVt + ot * 16 * 64, 64, lane, 0);
      v16bf bv1 = frag_b(sVt + ot * 16 * 64, 64, lane, 32);
      oacc[ot] = wmma_bf16(ap0, bv0, oacc[ot]);
      oacc[ot] = wmma_bf16(ap1, bv1, oacc[ot]);
    }
    __syncthreads();
  }

  // Normalize and write bf16, heads merged -> [B*S, D]
#pragma unroll
  for (int ot = 0; ot < 4; ++ot) {
    int col = h * HDq + ot * 16 + (lane & 15);
#pragma unroll
    for (int r = 0; r < 8; ++r) {
      int row = q0 + w * 16 + (lane >> 4) * 8 + r;
      float val = oacc[ot][r] / lrun[r];
      o[(size_t)(b * Sq + row) * Dq + col] = (bf16)val;
    }
  }
}

// ---------------------------------------------------------------------------
// Launch
// ---------------------------------------------------------------------------
extern "C" void kernel_launch(void* const* d_in, const int* in_sizes, int n_in,
                              void* d_out, int out_size, void* d_ws,
                              size_t ws_size, hipStream_t stream) {
  const float* x     = (const float*)d_in[0];  // [B,S,D]
  const float* W_qkv = (const float*)d_in[1];  // [D,3D]
  const float* b_qkv = (const float*)d_in[2];  // [3D]
  const float* W_out = (const float*)d_in[3];  // [D,D]
  const float* b_out = (const float*)d_in[4];  // [D]
  float* out = (float*)d_out;

  const int M = Bq * Sq;  // 8192

  // Workspace carve-up
  char* ws = (char*)d_ws;
  bf16* qkv   = (bf16*)(ws);                                  // 48 MB
  bf16* aout  = (bf16*)(ws + (size_t)M * 3 * Dq * 2);         // 16 MB
  bf16* x_bf  = (bf16*)(ws + (size_t)64  * 1024 * 1024);      // 16 MB
  bf16* Wq_t  = (bf16*)(ws + (size_t)80  * 1024 * 1024);      //  6 MB
  bf16* Wo_t  = (bf16*)(ws + (size_t)86  * 1024 * 1024);      //  2 MB

  // 0) One-shot conversions to bf16 (weights transposed -> [N][K])
  {
    int n = M * Dq;  // 8.38M
    cvt_f32_bf16<<<n / (256 * 4), 256, 0, stream>>>(x, x_bf, n);
    dim3 gq(3 * Dq / 32, Dq / 32);
    cvt_transpose_bf16<<<gq, 256, 0, stream>>>(W_qkv, Wq_t, Dq, 3 * Dq);
    dim3 go(Dq / 32, Dq / 32);
    cvt_transpose_bf16<<<go, 256, 0, stream>>>(W_out, Wo_t, Dq, Dq);
  }
  // 1) QKV projection -> bf16 qkv
  {
    dim3 grid(3 * Dq / 128, M / 128);
    gemm_bias_kernel<bf16>
        <<<grid, 256, 0, stream>>>(x_bf, Wq_t, b_qkv, qkv, M, 3 * Dq, Dq);
  }
  // 2) Attention
  {
    dim3 grid(Sq / 64, Bq * Hq);
    attn_kernel<<<grid, 128, 0, stream>>>(qkv, aout);
  }
  // 3) Output projection -> fp32 out
  {
    dim3 grid(Dq / 128, M / 128);
    gemm_bias_kernel<float>
        <<<grid, 256, 0, stream>>>(aout, Wo_t, b_out, out, M, Dq, Dq);
  }
}
